// QuantumGeoGNN_52656299049059
// MI455X (gfx1250) — matrix operand order repaired and checked by reading, not compiled, not run on hardware
//
#include <hip/hip_runtime.h>
#include <hip/hip_fp16.h>

#define NN      100000      // N_NODES
#define NE      3200000     // N_EDGES
#define NF      16          // N_FEAT
#define F2      32          // 2*N_FEAT
#define NL      3           // N_LAYERS
#define NG      128         // N_GRAPHS

typedef __attribute__((ext_vector_type(16))) _Float16 v16h;
typedef __attribute__((ext_vector_type(8)))  float    v8f;

union Frag { v16h h; uint4 u4[2]; int i32[8]; };

// CDNA5 hardware tanh (single trans-VALU op) instead of branchy libm tanhf.
__device__ __forceinline__ float fast_tanh(float x) {
#if __has_builtin(__builtin_amdgcn_tanhf)
  return __builtin_amdgcn_tanhf(x);
#else
  float r;
  asm("v_tanh_f32 %0, %1\n\tv_nop" : "=v"(r) : "v"(x));
  return r;
#endif
}

// v_cvt_pk_f16_f32: pack two f32 into one dword of two f16.
__device__ __forceinline__ int pack2h(float a, float b) {
  return __builtin_bit_cast(int, __builtin_amdgcn_cvt_pkrtz(a, b));
}

// Exchange register halves across the wave:
//   lo = {a.lanes[0:15],  b.lanes[0:15]}
//   hi = {a.lanes[16:31], b.lanes[16:31]}
// Fast path: single v_permlane16_swap_b32 (both outputs in one op).
// Fallback: xor-16 shuffle + per-half select.
__device__ __forceinline__ void halfswap(int a, int b, bool hiHalf,
                                         int& lo, int& hi) {
#if __has_builtin(__builtin_amdgcn_permlane16_swap)
  auto pr = __builtin_amdgcn_permlane16_swap((unsigned)a, (unsigned)b,
                                             false, false);
  lo = (int)pr[0];
  hi = (int)pr[1];
#else
  int ax = __shfl_xor(a, 16, 32);
  int bx = __shfl_xor(b, 16, 32);
  lo = hiHalf ? bx : a;
  hi = hiHalf ? b : ax;
#endif
}

// ---------------------------------------------------------------------------
// Kernel A: repack W_phi (3x32x32, row-major [l][k][out]) and W_meas (32x16)
// into f16 WMMA A-operand fragments, transposed: A_frag[m,k] = W[k, 16h+m].
// A layout (matches contiguous B): lane m (0..15) holds k=0..15;
// lane m+16 holds k=16..31; per-lane v16h element i maps to
// k = i + (lane>=16 ? 16 : 0).
// Fragments: 0..5 = layer l half h (f = 2l+h), 6 = W_meas. 512 halves each.
// ---------------------------------------------------------------------------
__global__ void qg_repack_kernel(const float* __restrict__ Wphi,
                                 const float* __restrict__ Wmeas,
                                 _Float16* __restrict__ wsA) {
  int t = blockIdx.x * blockDim.x + threadIdx.x;
  if (t >= 7 * 32 * 16) return;
  int i    = t & 15;
  int lane = (t >> 4) & 31;
  int frag = t >> 9;
  int m = lane & 15;
  int k = i + ((lane & 16) ? 16 : 0);
  float val;
  if (frag < 6) {
    int l = frag >> 1, h = frag & 1;
    val = Wphi[l * F2 * F2 + k * F2 + h * 16 + m];
  } else {
    val = Wmeas[k * NF + m];
  }
  wsA[frag * 512 + lane * 16 + i] = (_Float16)val;
}

// ---------------------------------------------------------------------------
// Kernel B: zero-fill scratch floats
// ---------------------------------------------------------------------------
__global__ void qg_zero_kernel(float* __restrict__ p, int n) {
  int t = blockIdx.x * blockDim.x + threadIdx.x;
  if (t < n) p[t] = 0.0f;
}

// ---------------------------------------------------------------------------
// Kernel C: per-edge quantum-MLP via WMMA + scatter-add into aggr.
// One wave per tile of 16 edges. Lane e (0..15) carries features k=0..15 of
// edge e; lane e+16 carries k=16..31 (contiguous k order, same as the A repack).
// ---------------------------------------------------------------------------
__global__ __launch_bounds__(256)
void qg_edge_kernel(const float* __restrict__ x,
                    const long long* __restrict__ ei,
                    const float* __restrict__ bphi,
                    const _Float16* __restrict__ wsA,
                    float* __restrict__ aggr) {
  const int lane = threadIdx.x & 31;
  const int wave = threadIdx.x >> 5;
  const int tile = blockIdx.x * (blockDim.x >> 5) + wave;
  const int e0   = tile * 16;
  if (e0 >= NE) return;                 // uniform across the wave
  const int e   = e0 + (lane & 15);
  const bool hi = (lane & 16) != 0;
  const int gb  = hi ? 8 : 0;

  const int rown = (int)ei[e];
  const int coln = (int)ei[(long long)NE + e];
  const int node = hi ? coln : rown;

  // ---- build B = Z^T for layer 0: 16 contiguous f32 of x[node] -> f16
  const float4* xp = (const float4*)(x + (long long)node * NF);
  float4 a = xp[0], b = xp[1], c = xp[2], d = xp[3];
  Frag B;
  B.i32[0] = pack2h(a.x, a.y);
  B.i32[1] = pack2h(a.z, a.w);
  B.i32[2] = pack2h(b.x, b.y);
  B.i32[3] = pack2h(b.z, b.w);
  B.i32[4] = pack2h(c.x, c.y);
  B.i32[5] = pack2h(c.z, c.w);
  B.i32[6] = pack2h(d.x, d.y);
  B.i32[7] = pack2h(d.z, d.w);

  // ---- three tanh layers: Z^T <- tanh(Wphi[l]^T * Z^T + b)
#pragma unroll
  for (int l = 0; l < NL; ++l) {
    Frag A0, A1;
    const uint4* p0 = (const uint4*)(wsA + (l * 2 + 0) * 512 + lane * 16);
    const uint4* p1 = (const uint4*)(wsA + (l * 2 + 1) * 512 + lane * 16);
    A0.u4[0] = p0[0]; A0.u4[1] = p0[1];
    A1.u4[0] = p1[0]; A1.u4[1] = p1[1];

    v8f c0 = {}, c1 = {};
    c0 = __builtin_amdgcn_wmma_f32_16x16x32_f16(false, A0.h, false, B.h,
                                                (short)0, c0, false, false);
    c1 = __builtin_amdgcn_wmma_f32_16x16x32_f16(false, A1.h, false, B.h,
                                                (short)0, c1, false, false);

    // bias: D_h vgpr v corresponds to output feature m' = 16h + v + gb
    const float* bl = bphi + l * F2;
    float4 b0a = *(const float4*)(bl + gb);
    float4 b0b = *(const float4*)(bl + gb + 4);
    float4 b1a = *(const float4*)(bl + 16 + gb);
    float4 b1b = *(const float4*)(bl + 16 + gb + 4);
    float bb0[8] = {b0a.x, b0a.y, b0a.z, b0a.w, b0b.x, b0b.y, b0b.z, b0b.w};
    float bb1[8] = {b1a.x, b1a.y, b1a.z, b1a.w, b1b.x, b1b.y, b1b.z, b1b.w};

    float t0[8], t1[8];
#pragma unroll
    for (int v = 0; v < 8; ++v) {
      t0[v] = fast_tanh(c0[v] + bb0[v]);
      t1[v] = fast_tanh(c1[v] + bb1[v]);
    }

    // Pack to f16 pairs, then C-layout -> B-layout register-half exchange.
    Frag Bn;
#pragma unroll
    for (int j = 0; j < 4; ++j) {
      int pk0 = pack2h(t0[2 * j], t0[2 * j + 1]);   // elements 2j,2j+1
      int pk1 = pack2h(t1[2 * j], t1[2 * j + 1]);
      halfswap(pk0, pk1, hi, Bn.i32[j], Bn.i32[j + 4]);
    }
    B = Bn;
  }

  // ---- measurement projection: msg^T[16,16] = tanh(Wmeas^T * Z^T)
  Frag Am;
  const uint4* pm = (const uint4*)(wsA + 6 * 512 + lane * 16);
  Am.u4[0] = pm[0]; Am.u4[1] = pm[1];
  v8f cm = {};
  cm = __builtin_amdgcn_wmma_f32_16x16x32_f16(false, Am.h, false, B.h,
                                              (short)0, cm, false, false);

  // lane e holds msg[e][0..7], lane e+16 holds msg[e][8..15]; scatter by row.
  float* ag = aggr + (long long)rown * NF + gb;
#pragma unroll
  for (int v = 0; v < 8; ++v) atomicAdd(ag + v, fast_tanh(cm[v]));
}

// ---------------------------------------------------------------------------
// Kernel D: node = concat(x, aggr) @ W_gamma + b_gamma; pool into graphs.
// ---------------------------------------------------------------------------
__global__ __launch_bounds__(256)
void qg_node_kernel(const float* __restrict__ x,
                    const float* __restrict__ aggr,
                    const long long* __restrict__ batch,
                    const float* __restrict__ Wg,
                    const float* __restrict__ bg,
                    float* __restrict__ sums,
                    float* __restrict__ cnts) {
  int n = blockIdx.x * blockDim.x + threadIdx.x;
  if (n >= NN) return;
  const float* xp = x + (long long)n * NF;
  const float* ap = aggr + (long long)n * NF;
  float acc = bg[0];
#pragma unroll
  for (int f = 0; f < NF; ++f) acc += xp[f] * Wg[f];
#pragma unroll
  for (int f = 0; f < NF; ++f) acc += ap[f] * Wg[NF + f];
  int g = (int)batch[n];
  atomicAdd(&sums[g], acc);
  atomicAdd(&cnts[g], 1.0f);
}

// ---------------------------------------------------------------------------
// Kernel E: out[g] = sums[g] / max(cnts[g], 1)
// ---------------------------------------------------------------------------
__global__ void qg_final_kernel(const float* __restrict__ sums,
                                const float* __restrict__ cnts,
                                float* __restrict__ out) {
  int g = blockIdx.x * blockDim.x + threadIdx.x;
  if (g < NG) out[g] = sums[g] / fmaxf(cnts[g], 1.0f);
}

// ---------------------------------------------------------------------------
extern "C" void kernel_launch(void* const* d_in, const int* in_sizes, int n_in,
                              void* d_out, int out_size, void* d_ws, size_t ws_size,
                              hipStream_t stream) {
  const float*     x     = (const float*)d_in[0];
  const long long* ei    = (const long long*)d_in[1];
  const long long* batch = (const long long*)d_in[2];
  const float*     Wphi  = (const float*)d_in[3];
  const float*     bphi  = (const float*)d_in[4];
  const float*     Wmeas = (const float*)d_in[5];
  const float*     Wg    = (const float*)d_in[6];
  const float*     bg    = (const float*)d_in[7];
  float*           out   = (float*)d_out;

  // workspace layout
  char*     ws   = (char*)d_ws;
  float*    aggr = (float*)ws;                              // NN*NF*4 = 6,400,000 B
  _Float16* wsA  = (_Float16*)(ws + (size_t)NN * NF * 4);   // 7*512*2 = 7,168 B
  float*    sums = (float*)(ws + (size_t)NN * NF * 4 + 7168);
  float*    cnts = sums + NG;                               // contiguous with sums

  // repack weights into WMMA A-fragments
  qg_repack_kernel<<<(7 * 32 * 16 + 255) / 256, 256, 0, stream>>>(Wphi, Wmeas, wsA);

  // zero aggr and pooled sums/counts (every call: deterministic)
  qg_zero_kernel<<<(NN * NF + 255) / 256, 256, 0, stream>>>(aggr, NN * NF);
  qg_zero_kernel<<<1, 256, 0, stream>>>(sums, 2 * NG);

  // per-edge WMMA MLP + scatter (16 edges per wave, 8 waves per block)
  const int tiles = NE / 16;             // 200,000
  const int waves_per_block = 256 / 32;  // 8
  qg_edge_kernel<<<(tiles + waves_per_block - 1) / waves_per_block, 256, 0, stream>>>(
      x, ei, bphi, wsA, aggr);

  // node gamma projection + graph pooling
  qg_node_kernel<<<(NN + 255) / 256, 256, 0, stream>>>(x, aggr, batch, Wg, bg,
                                                       sums, cnts);

  // finalize mean pool
  qg_final_kernel<<<1, NG, 0, stream>>>(sums, cnts, out);
}
